// canODE_attention_40922448396540
// MI455X (gfx1250) — compile-verified, matrix operand order checked by reading, not run on hardware
//
#include <hip/hip_runtime.h>
#include <math.h>

#define BDIM 64
#define NDIM 512
#define ROWS (BDIM * NDIM)   // 32768 (b,n) rows

typedef __attribute__((ext_vector_type(16))) __bf16    v16bf;
typedef __attribute__((ext_vector_type(8)))  float     v8f;
typedef __attribute__((ext_vector_type(4)))  unsigned  uint4x;
typedef __attribute__((ext_vector_type(4)))  int       v4i;
typedef __attribute__((address_space(1)))    v4i       as1_v4i;
typedef __attribute__((address_space(3)))    v4i       as3_v4i;

union Frag { uint4x u[2]; v16bf v; };

#if defined(__has_builtin)
#if __has_builtin(__builtin_amdgcn_global_load_async_to_lds_b128) && \
    __has_builtin(__builtin_amdgcn_s_wait_asynccnt)
#define HAVE_ASYNC_LDS 1
#endif
#endif

#ifdef HAVE_ASYNC_LDS
__device__ inline void async_copy_b128(const void* g, void* l) {
  __builtin_amdgcn_global_load_async_to_lds_b128((as1_v4i*)g, (as3_v4i*)l, 0, 0);
}
#endif

__device__ inline unsigned short f2bf(float f) {
  unsigned u = __float_as_uint(f);
  unsigned r = u + 0x7FFFu + ((u >> 16) & 1u);   // round-to-nearest-even
  return (unsigned short)(r >> 16);
}
__device__ inline float bf2f(unsigned short h) {
  return __uint_as_float(((unsigned)h) << 16);
}

// ---- one-time: transpose Wq/Wk to [out][in] bf16 (B-fragment friendly) ----
__global__ void transpose_w(const float* Wq, const float* Wk,
                            unsigned short* WqT, unsigned short* WkT) {
  int i = blockIdx.x * blockDim.x + threadIdx.x;   // 65536
  int o = i & 255, in = i >> 8;
  WqT[o * 256 + in] = f2bf(Wq[in * 256 + o]);
  WkT[o * 256 + in] = f2bf(Wk[in * 256 + o]);
}

// ---- one-time: embedding gather (slot0 = 0) + v_base = bv + e[1:]·Wv[1:] ----
__global__ void gather_embed(const long long* pos, const float* etab,
                             const float* Wv, const float* bv,
                             unsigned short* e_bf, float* v_base) {
  int i = blockIdx.x * blockDim.x + threadIdx.x;   // 32768
  long long p = pos[i];
  const float* row = etab + p * 255;
  unsigned short* dst = e_bf + (size_t)i * 256;
  dst[0] = 0;
  float acc = bv[0];
  for (int d = 1; d < 256; ++d) {
    float f = row[d - 1];
    dst[d] = f2bf(f);
    acc += f * Wv[d];
  }
  v_base[i] = acc;
}

// ---- one-time WMMA: q_base = [0|e]@Wq + bq ; k_base likewise (bf16 out) ----
__global__ __launch_bounds__(256) void base_mm(
    const unsigned short* e_bf,
    const unsigned short* WqT, const unsigned short* WkT,
    const float* bq, const float* bk,
    unsigned short* qbase, unsigned short* kbase) {
  int tid = threadIdx.x, wave = tid >> 5, lane = tid & 31;
  int lh = lane & 15, hi = lane >> 4;
  int r0 = blockIdx.x * 128 + wave * 16;
  int kb = hi ? 8 : 0;
  Frag A[8];
#pragma unroll
  for (int c = 0; c < 8; ++c) {
    const uint4x* p = (const uint4x*)(e_bf + (size_t)(r0 + lh) * 256 + c * 32 + kb);
    A[c].u[0] = p[0];
    A[c].u[1] = p[2];
  }
  for (int w = 0; w < 2; ++w) {
    const unsigned short* WT = w ? WkT : WqT;
    const float* bias = w ? bk : bq;
    unsigned short* outp = w ? kbase : qbase;
    for (int nt = 0; nt < 16; ++nt) {
      int ocol = nt * 16 + lh;
      v8f C = {};
#pragma unroll
      for (int c = 0; c < 8; ++c) {
        Frag Bf;
        const uint4x* p = (const uint4x*)(WT + (size_t)ocol * 256 + c * 32 + kb);
        Bf.u[0] = p[0];
        Bf.u[1] = p[2];
        C = __builtin_amdgcn_wmma_f32_16x16x32_bf16(false, A[c].v, false, Bf.v,
                                                    (short)0, C, false, false);
      }
      float bsv = bias[ocol];
#pragma unroll
      for (int j = 0; j < 8; ++j) {
        int row = r0 + j + hi * 8;
        outp[(size_t)row * 256 + ocol] = f2bf(C[j] + bsv);
      }
    }
  }
}

__global__ void init_state(const float* val, float* x, float* xe, float* y0) {
  int i = blockIdx.x * blockDim.x + threadIdx.x;
  float v = val[i];
  x[i] = v; xe[i] = v; y0[i] = v;
}

// ---- per-eval rank-1 update: q = q_base + x*Wq[0,:] (bf16), v likewise ----
__global__ void prep_qkv(const float* xe,
                         const unsigned short* qbase, const unsigned short* kbase,
                         const float* Wq, const float* Wk, const float* Wv,
                         const float* v_base,
                         unsigned short* qbf, unsigned short* kbf, float* vev) {
  int i = blockIdx.x * blockDim.x + threadIdx.x;   // 64*512*256
  int bn = i >> 8, d = i & 255;
  float xv = xe[bn];
  qbf[i] = f2bf(bf2f(qbase[i]) + xv * Wq[d]);      // Wq row 0
  kbf[i] = f2bf(bf2f(kbase[i]) + xv * Wk[d]);
  if (d == 0) vev[bn] = v_base[bn] + xv * Wv[0];
}

// ---- per-eval flash attention: fx[b,n] = softmax(q·kT/16) @ v ----
// Double-buffered async K staging: 16 key blocks of 32 keys (16 KB each).
__global__ __launch_bounds__(256) void attn(
    const unsigned short* qbf, const unsigned short* kbf,
    const float* vev, float* fx) {
  __shared__ __attribute__((aligned(16))) unsigned short kt[2][32 * 256];
  __shared__ float vt[2][32];
  int b = blockIdx.y;
  int tid = threadIdx.x, wave = tid >> 5, lane = tid & 31;
  int lh = lane & 15, hi = lane >> 4;
  int kb = hi ? 8 : 0;
  int r0 = blockIdx.x * 128 + wave * 16;           // 16 query rows per wave
  size_t qrow = ((size_t)b * 512 + r0 + lh) * 256;
  Frag A[8];
#pragma unroll
  for (int c = 0; c < 8; ++c) {
    const uint4x* p = (const uint4x*)(qbf + qrow + c * 32 + kb);
    A[c].u[0] = p[0]; A[c].u[1] = p[2];
  }
  float m[8], num[8], den[8];
#pragma unroll
  for (int j = 0; j < 8; ++j) { m[j] = -INFINITY; num[j] = 0.f; den[j] = 0.f; }

  const uint4x* ksrc = (const uint4x*)(kbf + (size_t)b * 512 * 256);
  // prologue: stage key block 0 into buffer 0
  {
    uint4x* dst = (uint4x*)kt[0];
#ifdef HAVE_ASYNC_LDS
    for (int i = tid; i < 512; i += 256) async_copy_b128(ksrc + i, dst + i);
#else
    for (int i = tid; i < 512; i += 256) dst[i] = ksrc[i];
#endif
    if (tid < 32) vt[0][tid] = vev[b * 512 + tid];
  }

#pragma unroll 1
  for (int blk = 0; blk < 16; ++blk) {
    int cur = blk & 1;
#ifdef HAVE_ASYNC_LDS
    __builtin_amdgcn_s_wait_asynccnt(0);
#endif
    __syncthreads();                               // buf[cur] ready for all waves
    if (blk + 1 < 16) {                            // prefetch next block into buf[1-cur]
      const uint4x* src = ksrc + (size_t)(blk + 1) * 512;
      uint4x* dst = (uint4x*)kt[1 - cur];
#ifdef HAVE_ASYNC_LDS
      for (int i = tid; i < 512; i += 256) async_copy_b128(src + i, dst + i);
#else
      for (int i = tid; i < 512; i += 256) dst[i] = src[i];
#endif
      if (tid < 32) vt[1 - cur][tid] = vev[b * 512 + (blk + 1) * 32 + tid];
    }
#pragma unroll 1
    for (int t = 0; t < 2; ++t) {                  // 2 key tiles of 16
      int keyl = t * 16 + lh;
      v8f C = {};
#pragma unroll
      for (int c = 0; c < 8; ++c) {                // K = 256 in 8 chunks of 32
        Frag Bf;
        const uint4x* p = (const uint4x*)(kt[cur] + keyl * 256 + c * 32 + kb);
        Bf.u[0] = p[0]; Bf.u[1] = p[2];
        C = __builtin_amdgcn_wmma_f32_16x16x32_bf16(false, A[c].v, false, Bf.v,
                                                    (short)0, C, false, false);
      }
      float vv = vt[cur][keyl];
#pragma unroll
      for (int j = 0; j < 8; ++j) {                // per-lane online softmax
        float s = C[j] * 0.0625f;                  // 1/sqrt(256)
        float nm = fmaxf(m[j], s);
        float e1 = __expf(s - nm);
        float c0 = __expf(m[j] - nm);
        num[j] = num[j] * c0 + e1 * vv;
        den[j] = den[j] * c0 + e1;
        m[j] = nm;
      }
    }
  }
  // merge the 16 key-column lanes once at the end
  float fxr[8];
#pragma unroll
  for (int j = 0; j < 8; ++j) {
    float nm = m[j];
    for (int off = 8; off; off >>= 1) nm = fmaxf(nm, __shfl_xor(nm, off, 16));
    float sc = __expf(m[j] - nm);
    float nn = num[j] * sc, dd = den[j] * sc;
    for (int off = 8; off; off >>= 1) { nn += __shfl_xor(nn, off, 16); dd += __shfl_xor(dd, off, 16); }
    fxr[j] = nn / dd;
  }
  if (lh == 0) {
    int base = b * 512 + r0 + hi * 8;
#pragma unroll
    for (int j = 0; j < 8; ++j) fx[base + j] = fxr[j];
  }
}

// ---- per-eval replicator dynamics + RK4 bookkeeping (one block per batch) ----
__global__ void combine(const float* tarr, const float* gate,
                        float* xe, const float* fx,
                        float* x, float* acc, int stage, float* y1) {
  __shared__ float sred[256];
  int b = blockIdx.x, tid = threadIdx.x;
  int i0 = b * 512 + tid, i1 = i0 + 256;
  float xe0 = xe[i0], xe1 = xe[i1];
  float f0 = fx[i0], f1 = fx[i1];
  sred[tid] = xe0 * f0 + xe1 * f1;
  __syncthreads();
  for (int s = 128; s; s >>= 1) {
    if (tid < s) sred[tid] += sred[tid + s];
    __syncthreads();
  }
  float dot = sred[0];
  float g = gate[0];
  float h = (tarr[1] - tarr[0]) * 0.5f;            // h_interval / SUBSTEPS
  float k0 = g * xe0 * (f0 - dot);
  float k1 = g * xe1 * (f1 - dot);
  float w = (stage == 1 || stage == 2) ? 2.f : 1.f;
  float a0 = (stage == 0) ? w * k0 : acc[i0] + w * k0;
  float a1 = (stage == 0) ? w * k1 : acc[i1] + w * k1;
  acc[i0] = a0; acc[i1] = a1;
  if (stage < 3) {
    float c = (stage == 2) ? h : 0.5f * h;
    xe[i0] = x[i0] + c * k0;
    xe[i1] = x[i1] + c * k1;
  } else {
    float h6 = h * (1.f / 6.f);
    float xn0 = x[i0] + h6 * a0;
    float xn1 = x[i1] + h6 * a1;
    x[i0] = xn0; x[i1] = xn1;
    xe[i0] = xn0; xe[i1] = xn1;
    if (y1) { y1[i0] = xn0; y1[i1] = xn1; }
  }
}

extern "C" void kernel_launch(void* const* d_in, const int* in_sizes, int n_in,
                              void* d_out, int out_size, void* d_ws, size_t ws_size,
                              hipStream_t stream) {
  (void)in_sizes; (void)n_in; (void)out_size; (void)ws_size;
  const float*     t    = (const float*)d_in[0];
  const float*     val  = (const float*)d_in[1];
  const long long* pos  = (const long long*)d_in[2];
  const float*     etab = (const float*)d_in[3];
  const float*     Wq   = (const float*)d_in[4];
  const float*     bq   = (const float*)d_in[5];
  const float*     Wk   = (const float*)d_in[6];
  const float*     bk   = (const float*)d_in[7];
  const float*     Wv   = (const float*)d_in[8];
  const float*     bv   = (const float*)d_in[9];
  const float*     gate = (const float*)d_in[10];
  float* out = (float*)d_out;

  char* ws = (char*)d_ws;
  size_t o = 0;
  auto alloc = [&](size_t bytes) {
    char* p = ws + o; o += (bytes + 255) & ~(size_t)255; return p;
  };
  unsigned short* e_bf  = (unsigned short*)alloc((size_t)ROWS * 256 * 2);
  unsigned short* qbase = (unsigned short*)alloc((size_t)ROWS * 256 * 2);
  unsigned short* kbase = (unsigned short*)alloc((size_t)ROWS * 256 * 2);
  unsigned short* qbf   = (unsigned short*)alloc((size_t)ROWS * 256 * 2);
  unsigned short* kbf   = (unsigned short*)alloc((size_t)ROWS * 256 * 2);
  unsigned short* WqT   = (unsigned short*)alloc(256 * 256 * 2);
  unsigned short* WkT   = (unsigned short*)alloc(256 * 256 * 2);
  float* v_base = (float*)alloc(ROWS * 4);
  float* vev    = (float*)alloc(ROWS * 4);
  float* fx     = (float*)alloc(ROWS * 4);
  float* x      = (float*)alloc(ROWS * 4);
  float* xe     = (float*)alloc(ROWS * 4);
  float* acc    = (float*)alloc(ROWS * 4);

  transpose_w<<<256, 256, 0, stream>>>(Wq, Wk, WqT, WkT);
  gather_embed<<<128, 256, 0, stream>>>(pos, etab, Wv, bv, e_bf, v_base);
  base_mm<<<256, 256, 0, stream>>>(e_bf, WqT, WkT, bq, bk, qbase, kbase);
  init_state<<<128, 256, 0, stream>>>(val, x, xe, out);

  for (int sub = 0; sub < 2; ++sub) {
    for (int stage = 0; stage < 4; ++stage) {
      prep_qkv<<<ROWS, 256, 0, stream>>>(xe, qbase, kbase, Wq, Wk, Wv,
                                         v_base, qbf, kbf, vev);
      attn<<<dim3(4, 64), 256, 0, stream>>>(qbf, kbf, vev, fx);
      float* y1 = (sub == 1 && stage == 3) ? (out + ROWS) : nullptr;
      combine<<<64, 256, 0, stream>>>(t, gate, xe, fx, x, acc, stage, y1);
    }
  }
}
